// ScanAttention_82824149336915
// MI455X (gfx1250) — matrix operand order repaired and checked by reading, not compiled, not run on hardware
//
#include <hip/hip_runtime.h>
#include <hip/hip_bf16.h>

// FlashAttention-2 forward for B=1, H=16, Q=K=4096, D=64 (fp32 in/out),
// computed with v_wmma_f32_16x16x32_f16 (f16 operands, f32 accumulation).
// Softmax in log2 domain (scale * log2e folded into Q conversion).
// Row sums of P computed on the matrix pipe via an all-ones B fragment.
// 128-key tiles amortize the softmax epilogue; uniform-branch guards skip
// the mask and max-rescale work when dynamically dead.

typedef __attribute__((ext_vector_type(16))) _Float16 v16h;
typedef __attribute__((ext_vector_type(8)))  _Float16 v8h;
typedef __attribute__((ext_vector_type(4)))  _Float16 v4h;
typedef __attribute__((ext_vector_type(8)))  float    v8f;

#define HQ    16
#define QLEN  4096
#define KLEN  4096
#define DDIM  64

#define BQ       128   // q rows per block (8 waves x 16 rows)
#define BK       128   // keys per tile
#define NKT      (BK / 16)        // 8 N-tiles in S
#define NC       (BK / 32)        // 4 K-chunks in PV
#define KSTRIDE  80    // ldsK row stride in halves (160B -> 32B-aligned v16h loads)
#define VSTRIDE  144   // ldsV row stride in halves (288B, 128 keys + pad)
#define PSTRIDE  144   // ldsP row stride in halves (288B, 128 cols + pad)

__global__ __launch_bounds__(256)
void ScanAttention_82824149336915_kernel(const float* __restrict__ qs,
                                         const float* __restrict__ ks,
                                         const float* __restrict__ vs,
                                         const unsigned char* __restrict__ mask,
                                         float* __restrict__ out)
{
    __shared__ __attribute__((aligned(64))) _Float16 ldsK[BK * KSTRIDE];      // [key][d]
    __shared__ __attribute__((aligned(64))) _Float16 ldsV[DDIM * VSTRIDE];    // [d][key] (transposed)
    __shared__ __attribute__((aligned(64))) _Float16 ldsP[8 * 16 * PSTRIDE];  // per-wave P scratch

    const int h    = blockIdx.y;
    const int q0   = blockIdx.x * BQ;
    const int tid  = threadIdx.x;
    const int wave = tid >> 5;
    const int lane = tid & 31;
    const int n    = lane & 15;   // N column of B/C fragments
    const int hi   = lane >> 4;   // lane half
    const int rb   = hi * 8;      // C-fragment row base
    const int c0   = hi * 8;      // unused; A-fragments also use kb
    const int kb   = hi * 16;     // B-fragment K sub-offset

    // 1/sqrt(64) * log2(e): scores land directly in log2 domain.
    const float scale = 0.125f * 1.44269504088896340736f;

    // Per-head base pointers (32-bit offsets from here on).
    const float4* kg4 = (const float4*)(ks + (size_t)h * KLEN * DDIM);
    const float4* vg4 = (const float4*)(vs + (size_t)h * KLEN * DDIM);

    // Fixed per-lane staging bases: slot idx = tid + p*256, row = idx>>4,
    // col4 = idx&15, so row*16+col4 == idx and the p-dimension is a pure
    // immediate offset for both the global loads and the LDS stores.
    const int row0 = tid >> 4;
    const int col4 = tid & 15;
    _Float16* kst = &ldsK[row0 * KSTRIDE + col4 * 4];
    _Float16* vst = &ldsV[(col4 * 4) * VSTRIDE + row0];

    // ---- Load this wave's 16 Q rows as A-fragments (2 chunks of K=32) ----
    const int    qrow = q0 + wave * 16 + n;
    const float* qp   = qs + ((size_t)h * QLEN + qrow) * DDIM;
    v16h aq[2];
    #pragma unroll
    for (int c = 0; c < 2; ++c) {
        #pragma unroll
        for (int j = 0; j < 8; ++j) {
            aq[c][j]     = (_Float16)(qp[c * 32 + kb + j]      * scale);
            aq[c][j + 8] = (_Float16)(qp[c * 32 + kb + 8 + j] * scale);
        }
    }

    // Constant all-ones B fragment: row-sum GEMM operand.
    v16h hones;
    #pragma unroll
    for (int j = 0; j < 16; ++j) hones[j] = (_Float16)1.0f;

    v8f   oacc[4];
    float mrow[8], lrow[8];
    #pragma unroll
    for (int t = 0; t < 4; ++t) oacc[t] = {};
    #pragma unroll
    for (int r = 0; r < 8; ++r) { mrow[r] = -__builtin_inff(); lrow[r] = 0.0f; }

    for (int kt = 0; kt < KLEN; kt += BK) {
        __syncthreads();  // protect K/V LDS from previous iteration's readers

        // ---- Cooperative stage: K tile (row-major f16), V tile (transposed f16) ----
        {
            const float4* kgt = kg4 + (size_t)kt * (DDIM / 4);
            const float4* vgt = vg4 + (size_t)kt * (DDIM / 4);
            #pragma unroll
            for (int p = 0; p < 8; ++p) {
                float4 kv = kgt[tid + p * 256];
                v4h kp;
                kp[0] = (_Float16)kv.x; kp[1] = (_Float16)kv.y;
                kp[2] = (_Float16)kv.z; kp[3] = (_Float16)kv.w;
                *(v4h*)(kst + p * 16 * KSTRIDE) = kp;   // one ds_store_b64
                float4 vv = vgt[tid + p * 256];
                vst[0 * VSTRIDE + p * 16] = (_Float16)vv.x;
                vst[1 * VSTRIDE + p * 16] = (_Float16)vv.y;
                vst[2 * VSTRIDE + p * 16] = (_Float16)vv.z;
                vst[3 * VSTRIDE + p * 16] = (_Float16)vv.w;
            }
        }
        __syncthreads();

        // Prefetch next K/V tile: 256 threads x 128B == one 32KB tile each.
        if (kt + BK < KLEN) {
            const char* nk = (const char*)kg4 + (size_t)(kt + BK) * DDIM * 4;
            const char* nv = (const char*)vg4 + (size_t)(kt + BK) * DDIM * 4;
            __builtin_prefetch(nk + tid * 128, 0, 1);
            __builtin_prefetch(nk + tid * 128 + 64, 0, 1);
            __builtin_prefetch(nv + tid * 128, 0, 1);
            __builtin_prefetch(nv + tid * 128 + 64, 0, 1);
        }

        // Hoist mask bytes so the loads retire under the QK WMMAs.
        unsigned char mb[NKT];
        #pragma unroll
        for (int nt = 0; nt < NKT; ++nt) mb[nt] = mask[kt + nt * 16 + n];

        // ---- S = (Q * scale) K^T : 8 N-tiles x 2 K-chunks, pipelined B loads ----
        v8f  s[NKT];
        v16h b0 = *(const v16h*)&ldsK[n * KSTRIDE + kb];
        v16h b1 = *(const v16h*)&ldsK[n * KSTRIDE + 32 + kb];
        #pragma unroll
        for (int nt = 0; nt < NKT; ++nt) {
            v16h nb0, nb1;
            if (nt < NKT - 1) {
                nb0 = *(const v16h*)&ldsK[((nt + 1) * 16 + n) * KSTRIDE + kb];
                nb1 = *(const v16h*)&ldsK[((nt + 1) * 16 + n) * KSTRIDE + 32 + kb];
            }
            v8f acc = {};
            acc = __builtin_amdgcn_wmma_f32_16x16x32_f16(
                      false, aq[0], false, b0, (short)0, acc, false, false);
            acc = __builtin_amdgcn_wmma_f32_16x16x32_f16(
                      false, aq[1], false, b1, (short)0, acc, false, false);
            s[nt] = acc;
            b0 = nb0; b1 = nb1;
        }

        // Key mask (reference: scores -> -inf where !mask). Uniform-branch
        // guarded: no dynamic cndmask work when the mask is all-true.
        {
            bool anym = false;
            #pragma unroll
            for (int nt = 0; nt < NKT; ++nt) anym |= (mb[nt] == 0);
            if (__any(anym)) {
                #pragma unroll
                for (int nt = 0; nt < NKT; ++nt) {
                    if (!mb[nt]) {
                        #pragma unroll
                        for (int r = 0; r < 8; ++r) s[nt][r] = -__builtin_inff();
                    }
                }
            }
        }

        // ---- Online softmax (log2 domain), one epilogue per 128 keys ----
        float tmax[8];
        #pragma unroll
        for (int r = 0; r < 8; ++r) {
            float t0 = fmaxf(fmaxf(s[0][r], s[1][r]), fmaxf(s[2][r], s[3][r]));
            float t1 = fmaxf(fmaxf(s[4][r], s[5][r]), fmaxf(s[6][r], s[7][r]));
            tmax[r]  = fmaxf(t0, t1);
        }
        #pragma unroll
        for (int x = 1; x < 16; x <<= 1) {
            #pragma unroll
            for (int r = 0; r < 8; ++r)
                tmax[r] = fmaxf(tmax[r], __shfl_xor(tmax[r], x, 32));
        }

        // Rescale path only when some row max actually advanced (FA-2 skip).
        {
            bool chg = false;
            #pragma unroll
            for (int r = 0; r < 8; ++r) chg |= (tmax[r] > mrow[r]);
            if (__any(chg)) {
                float alpha[8];
                #pragma unroll
                for (int r = 0; r < 8; ++r) {
                    float mn = fmaxf(mrow[r], tmax[r]);
                    alpha[r] = __builtin_amdgcn_exp2f(mrow[r] - mn);  // exp2(-inf)=0 first pass
                    mrow[r]  = mn;
                    lrow[r] *= alpha[r];
                }
                #pragma unroll
                for (int t = 0; t < 4; ++t)
                    #pragma unroll
                    for (int r = 0; r < 8; ++r) oacc[t][r] *= alpha[r];
            }
        }

        // P = exp2(S - m) -> f16 -> per-wave LDS scratch (row sums come from WMMA below)
        #pragma unroll
        for (int nt = 0; nt < NKT; ++nt) {
            #pragma unroll
            for (int r = 0; r < 8; ++r) {
                float p = __builtin_amdgcn_exp2f(s[nt][r] - mrow[r]);
                ldsP[(wave * 16 + rb + r) * PSTRIDE + nt * 16 + n] = (_Float16)p;
            }
        }

        __syncthreads();  // P scratch visible to all lanes of the wave

        // ---- O += P V  and  l += P * ones : 4 K-chunks, pipelined V^T B loads ----
        v8f lacc = {};
        #pragma unroll
        for (int c = 0; c < NC; ++c) {
            const _Float16* prow = &ldsP[(wave * 16 + n) * PSTRIDE + c * 32];
            v8h lo  = *(const v8h*)(prow + kb);
            v8h hi8 = *(const v8h*)(prow + kb + 8);
            v16h ap;
            #pragma unroll
            for (int j = 0; j < 8; ++j) { ap[j] = lo[j]; ap[j + 8] = hi8[j]; }

            // Row-sum GEMM on the matrix pipe (replaces shfl-add reduction).
            lacc = __builtin_amdgcn_wmma_f32_16x16x32_f16(
                       false, ap, false, hones, (short)0, lacc, false, false);

            v16h bv = *(const v16h*)&ldsV[n * VSTRIDE + c * 32 + kb];
            #pragma unroll
            for (int t = 0; t < 4; ++t) {
                v16h nb;
                if (t < 3)
                    nb = *(const v16h*)&ldsV[((t + 1) * 16 + n) * VSTRIDE + c * 32 + kb];
                oacc[t] = __builtin_amdgcn_wmma_f32_16x16x32_f16(
                              false, ap, false, bv, (short)0, oacc[t], false, false);
                bv = nb;
            }
        }

        #pragma unroll
        for (int r = 0; r < 8; ++r) lrow[r] += lacc[r];
    }

    // ---- Normalize and write out (f32) ----
    float* op = out + ((size_t)h * QLEN + q0 + wave * 16) * DDIM;
    #pragma unroll
    for (int r = 0; r < 8; ++r) {
        float inv = 1.0f / lrow[r];
        #pragma unroll
        for (int t = 0; t < 4; ++t)
            op[(rb + r) * DDIM + t * 16 + n] = oacc[t][r] * inv;
    }
}

extern "C" void kernel_launch(void* const* d_in, const int* in_sizes, int n_in,
                              void* d_out, int out_size, void* d_ws, size_t ws_size,
                              hipStream_t stream) {
    (void)in_sizes; (void)n_in; (void)d_ws; (void)ws_size; (void)out_size;
    const float*         qs   = (const float*)d_in[0];
    const float*         ks   = (const float*)d_in[1];
    const float*         vs   = (const float*)d_in[2];
    const unsigned char* mask = (const unsigned char*)d_in[3];  // jnp bool_: 1 byte/elem
    float*               out  = (float*)d_out;

    dim3 grid(QLEN / BQ, HQ, 1);   // 32 q-tiles x 16 heads
    ScanAttention_82824149336915_kernel<<<grid, 256, 0, stream>>>(qs, ks, vs, mask, out);
}
